// Attention_31447750542006
// MI455X (gfx1250) — matrix operand order repaired and checked by reading, not compiled, not run on hardware
//
#include <hip/hip_runtime.h>
#include <hip/hip_bf16.h>
#include <math.h>

// ---------------------------------------------------------------------------
// MI455X attention forward. All matmuls (QKV proj, Q·K^T, P·V, out-proj) go
// through v_wmma_f32_16x16x32_bf16 (f32 accumulate). Flash-attention online
// softmax with DPP16 butterfly reductions; V fragments hoisted above softmax
// and next K/V tiles prefetched so matrix pipes don't stall on L2.
// ---------------------------------------------------------------------------

typedef __bf16 bf16_t;
typedef __attribute__((ext_vector_type(16))) __bf16 v16bf;
typedef __attribute__((ext_vector_type(8)))  __bf16 v8bf;
typedef __attribute__((ext_vector_type(4)))  __bf16 v4bf;
typedef __attribute__((ext_vector_type(8)))  float  v8f;

#define DIM   1024
#define NH    16
#define DH    64
#define BATCH 2
#define SEQ   2048
#define MTOK  (BATCH * SEQ)   // 4096 tokens

__device__ __forceinline__ bf16_t f2bf(float f) {
  unsigned u = __float_as_uint(f);
  u += 0x7FFFu + ((u >> 16) & 1u);          // round-to-nearest-even
  unsigned short h = (unsigned short)(u >> 16);
  return __builtin_bit_cast(bf16_t, h);
}
__device__ __forceinline__ float bf2f(bf16_t b) {
  unsigned short h = __builtin_bit_cast(unsigned short, b);
  return __uint_as_float((unsigned)h << 16);
}

__device__ __forceinline__ v8f wmma_bf16(v16bf a, v16bf b, v8f c) {
  // (neg_a, A, neg_b, B, c_mod, C, reuse_a, reuse_b)
  return __builtin_amdgcn_wmma_f32_16x16x32_bf16(false, a, false, b, (short)0, c,
                                                 false, false);
}

// 16-lane butterfly reductions in pure VALU via DPP16.
// Masks {1,2,7,15}: quad_perm(1,0,3,2)=0xB1, quad_perm(2,3,0,1)=0x4E,
// row_half_mirror=0x141, row_mirror=0x140 -- generate the full 16-lane group.
__device__ __forceinline__ float dpp_max16(float v) {
  v = fmaxf(v, __int_as_float(__builtin_amdgcn_mov_dpp(__float_as_int(v), 0xB1,  0xF, 0xF, true)));
  v = fmaxf(v, __int_as_float(__builtin_amdgcn_mov_dpp(__float_as_int(v), 0x4E,  0xF, 0xF, true)));
  v = fmaxf(v, __int_as_float(__builtin_amdgcn_mov_dpp(__float_as_int(v), 0x141, 0xF, 0xF, true)));
  v = fmaxf(v, __int_as_float(__builtin_amdgcn_mov_dpp(__float_as_int(v), 0x140, 0xF, 0xF, true)));
  return v;
}
__device__ __forceinline__ float dpp_sum16(float v) {
  v += __int_as_float(__builtin_amdgcn_mov_dpp(__float_as_int(v), 0xB1,  0xF, 0xF, true));
  v += __int_as_float(__builtin_amdgcn_mov_dpp(__float_as_int(v), 0x4E,  0xF, 0xF, true));
  v += __int_as_float(__builtin_amdgcn_mov_dpp(__float_as_int(v), 0x141, 0xF, 0xF, true));
  v += __int_as_float(__builtin_amdgcn_mov_dpp(__float_as_int(v), 0x140, 0xF, 0xF, true));
  return v;
}

// A-matrix 16x32 bf16 fragment (ISA 7.12.2): lane m=lane&15 holds
// K = {0..7, 16..23} (lanes 0-15) / {8..15, 24..31} (lanes 16-31).
__device__ __forceinline__ v16bf load_frag_a(const bf16_t* __restrict__ base,
                                             int ld, int lane) {
  const int r  = lane & 15;
  const int k0 = (lane >> 4) << 3;          // 0 or 8
  const bf16_t* p = base + (size_t)r * ld + k0;
  v8bf lo = *(const v8bf*)(p);
  v8bf hi = *(const v8bf*)(p + 16);
  return __builtin_shufflevector(lo, hi, 0, 1, 2, 3, 4, 5, 6, 7,
                                 8, 9, 10, 11, 12, 13, 14, 15);
}

// B-matrix 32x16 bf16 fragment: lane n=lane&15 holds K={0..7,16..23} (lanes 0-15) /
// K={8..15,24..31} (lanes 16-31), same order as A. base is row-major over [N, K]
// (i.e. B^T in memory), ld = K stride.
__device__ __forceinline__ v16bf load_frag_b(const bf16_t* __restrict__ base,
                                             int ld, int lane) {
  const int r  = lane & 15;
  const int k0 = (lane >> 4) << 3;          // 0 or 8
  const bf16_t* p = base + (size_t)r * ld + k0;
  v8bf lo = *(const v8bf*)(p);
  v8bf hi = *(const v8bf*)(p + 16);
  return __builtin_shufflevector(lo, hi, 0, 1, 2, 3, 4, 5, 6, 7,
                                 8, 9, 10, 11, 12, 13, 14, 15);
}

// ---------------------------------------------------------------------------
__global__ __launch_bounds__(256) void cvt_f32_bf16(const float* __restrict__ src,
                                                    bf16_t* __restrict__ dst,
                                                    int n4) {
  int i = blockIdx.x * blockDim.x + threadIdx.x;
  if (i >= n4) return;
  float4 f = ((const float4*)src)[i];
  v4bf o;
  o[0] = f2bf(f.x); o[1] = f2bf(f.y); o[2] = f2bf(f.z); o[3] = f2bf(f.w);
  ((v4bf*)dst)[i] = o;
}

// ---------------------------------------------------------------------------
// C[M,N] = A[M,K] @ W[N,K]^T  (both row-major bf16), f32 accumulate.
// Block = 256 threads = 8 waves; wave tile 64x64 (4x4 WMMA -> 16 wmma per
// 16 b128 loads per k-step); block tile 128x256.
template <bool F32OUT>
__global__ __launch_bounds__(256) void gemm_xwT(const bf16_t* __restrict__ A,
                                                const bf16_t* __restrict__ W,
                                                void* __restrict__ Cout,
                                                int M, int N, int K) {
  const int lane = threadIdx.x & 31;
  const int wave = threadIdx.x >> 5;
  const int row0 = blockIdx.y * 128 + (wave >> 2) * 64;
  const int col0 = blockIdx.x * 256 + (wave & 3) * 64;

  v8f acc[4][4] = {};

  for (int kk = 0; kk < K; kk += 32) {
    v16bf a[4], b[4];
#pragma unroll
    for (int i = 0; i < 4; ++i)
      a[i] = load_frag_a(A + (size_t)(row0 + 16 * i) * K + kk, K, lane);
#pragma unroll
    for (int j = 0; j < 4; ++j)
      b[j] = load_frag_b(W + (size_t)(col0 + 16 * j) * K + kk, K, lane);
#pragma unroll
    for (int i = 0; i < 4; ++i)
#pragma unroll
      for (int j = 0; j < 4; ++j)
        acc[i][j] = wmma_bf16(a[i], b[j], acc[i][j]);
  }

  // C layout: reg r -> row r (lanes 0-15) / r+8 (lanes 16-31); lane&15 -> col.
  const int cn = lane & 15;
  const int mr = (lane >> 4) << 3;
#pragma unroll
  for (int i = 0; i < 4; ++i) {
#pragma unroll
    for (int j = 0; j < 4; ++j) {
#pragma unroll
      for (int r = 0; r < 8; ++r) {
        const size_t m = (size_t)(row0 + 16 * i + mr + r);
        const size_t n = (size_t)(col0 + 16 * j + cn);
        if constexpr (F32OUT) {
          ((float*)Cout)[m * N + n] = acc[i][j][r];
        } else {
          ((bf16_t*)Cout)[m * N + n] = f2bf(acc[i][j][r]);
        }
      }
    }
  }
}

// ---------------------------------------------------------------------------
// RoPE + relayout: src [B,S,H*Dh] -> dst [B,H,S,Dh]. One thread per rot-pair.
__global__ __launch_bounds__(256) void rope_relayout(const bf16_t* __restrict__ src,
                                                     bf16_t* __restrict__ dst) {
  const int idx = blockIdx.x * blockDim.x + threadIdx.x;   // 2^21 threads
  const int i = idx & 31;                  // pair index 0..31
  const int s = (idx >> 5) & (SEQ - 1);
  const int h = (idx >> 16) & (NH - 1);
  const int b = idx >> 20;
  const size_t si = ((size_t)(b * SEQ + s) * NH + h) * DH;
  const float x1 = bf2f(src[si + i]);
  const float x2 = bf2f(src[si + i + 32]);
  // inv_freq = 10000^(-i/32) = exp(-i * ln(10000)/32)
  const float f = (float)s * __expf(-(float)i * 0.2878231366242557f);
  float sn, c;
  __sincosf(f, &sn, &c);
  const size_t di = ((size_t)((b * NH + h) * SEQ + s)) * DH;
  dst[di + i]      = f2bf(x1 * c - x2 * sn);
  dst[di + i + 32] = f2bf(x2 * c + x1 * sn);
}

// V relayout + transpose: src [B,S,H*Dh] -> dst [B,H,Dh,S].
__global__ __launch_bounds__(256) void v_transpose(const bf16_t* __restrict__ src,
                                                   bf16_t* __restrict__ dst) {
  const int idx = blockIdx.x * blockDim.x + threadIdx.x;   // 2^22 threads
  const int d = idx & (DH - 1);
  const int s = (idx >> 6) & (SEQ - 1);
  const int h = (idx >> 17) & (NH - 1);
  const int b = idx >> 21;
  dst[((size_t)((b * NH + h) * DH + d)) * SEQ + s] =
      src[((size_t)(b * SEQ + s) * NH + h) * DH + d];
}

// ---------------------------------------------------------------------------
// Flash attention. Grid = (B*H, S/64); block = 128 threads = 4 waves, each
// wave owns a 16-row q tile; 64 keys per inner iteration (8 score WMMA +
// 8 PV WMMA), softmax reductions in DPP16. V fragments are loaded before the
// softmax (independent work in flight), next K/V tiles are prefetched.
__global__ __launch_bounds__(128) void flash_attn(const bf16_t* __restrict__ Qr,
                                                  const bf16_t* __restrict__ Kr,
                                                  const bf16_t* __restrict__ Vt,
                                                  bf16_t* __restrict__ Ob) {
  __shared__ bf16_t p_lds[4][16 * 64];      // per-wave P tile (16 q x 64 k)

  const int lane  = threadIdx.x & 31;
  const int wave  = threadIdx.x >> 5;
  const int bh    = blockIdx.x;             // b*NH + h
  const int q0    = blockIdx.y * 64 + wave * 16;
  const int cn    = lane & 15;
  const int khalf = lane >> 4;
  const int fr    = cn;                     // fragment row index this lane reads
  const int fk    = khalf << 4;             // V prefetch k offset (0/16)

  const bf16_t* Qh = Qr + (size_t)bh * SEQ * DH;   // [S, Dh]
  const bf16_t* Kh = Kr + (size_t)bh * SEQ * DH;   // [S, Dh]
  const bf16_t* Vh = Vt + (size_t)bh * DH * SEQ;   // [Dh, S]

  // Q fragments (A-matrix), d-halves 0..31 and 32..63, reused all iterations.
  const v16bf aq0 = load_frag_a(Qh + (size_t)q0 * DH, DH, lane);
  const v16bf aq1 = load_frag_a(Qh + (size_t)q0 * DH + 32, DH, lane);

  v8f o[4] = {};                            // O tile 16 x 64 (4 C-frags)
  float rm[8], rl[8];
#pragma unroll
  for (int r = 0; r < 8; ++r) { rm[r] = -1e30f; rl[r] = 0.0f; }

  bf16_t* pw = &p_lds[wave][0];

  for (int kt = 0; kt < SEQ / 64; ++kt) {
    // ---- Scores S = Q K^T * 1/sqrt(Dh): four 16x16 key tiles -------------
    const bf16_t* Kb = Kh + (size_t)(kt * 64) * DH;
    v8f s0 = {}, s1 = {}, s2 = {}, s3 = {};
    {
      v16bf b0, b1;
      b0 = load_frag_b(Kb, DH, lane);
      b1 = load_frag_b(Kb + 32, DH, lane);
      s0 = wmma_bf16(aq0, b0, s0); s0 = wmma_bf16(aq1, b1, s0);
      b0 = load_frag_b(Kb + 16 * DH, DH, lane);
      b1 = load_frag_b(Kb + 16 * DH + 32, DH, lane);
      s1 = wmma_bf16(aq0, b0, s1); s1 = wmma_bf16(aq1, b1, s1);
      b0 = load_frag_b(Kb + 32 * DH, DH, lane);
      b1 = load_frag_b(Kb + 32 * DH + 32, DH, lane);
      s2 = wmma_bf16(aq0, b0, s2); s2 = wmma_bf16(aq1, b1, s2);
      b0 = load_frag_b(Kb + 48 * DH, DH, lane);
      b1 = load_frag_b(Kb + 48 * DH + 32, DH, lane);
      s3 = wmma_bf16(aq0, b0, s3); s3 = wmma_bf16(aq1, b1, s3);
    }

    // ---- V fragments for this tile: issue now, consume after softmax -----
    const bf16_t* Vb = Vh + kt * 64;
    v16bf vf[4][2];
#pragma unroll
    for (int t = 0; t < 4; ++t) {
      vf[t][0] = load_frag_b(Vb + (size_t)(t * 16) * SEQ, SEQ, lane);
      vf[t][1] = load_frag_b(Vb + (size_t)(t * 16) * SEQ + 32, SEQ, lane);
    }

    // ---- Prefetch next iteration's K/V rows (speculative, L2->L0) --------
    // K rows are 128B-aligned; one prefetch per fragment row covers both
    // 16B chunks this lane will load. global_prefetch_b8, no LOADcnt cost.
    {
      const bf16_t* Kn = Kb + (size_t)64 * DH;
      const bf16_t* Vn = Vb + 64;
#pragma unroll
      for (int t = 0; t < 4; ++t) {
        __builtin_prefetch(Kn + (size_t)(t * 16 + fr) * DH, 0, 3);
        __builtin_prefetch(Vn + (size_t)(t * 16 + fr) * SEQ + fk, 0, 3);
      }
    }

    // ---- Online softmax per q-row (row spread across 16 lanes of a half) -
#pragma unroll
    for (int r = 0; r < 8; ++r) {
      const float v0 = s0[r] * 0.125f;      // 1/sqrt(64)
      const float v1 = s1[r] * 0.125f;
      const float v2 = s2[r] * 0.125f;
      const float v3 = s3[r] * 0.125f;
      const float mx = dpp_max16(fmaxf(fmaxf(v0, v1), fmaxf(v2, v3)));
      const float mn   = fmaxf(rm[r], mx);
      const float corr = __expf(rm[r] - mn);
      const float p0   = __expf(v0 - mn);
      const float p1   = __expf(v1 - mn);
      const float p2   = __expf(v2 - mn);
      const float p3   = __expf(v3 - mn);
      const float rs   = dpp_sum16((p0 + p1) + (p2 + p3));
      rl[r] = rl[r] * corr + rs;
      rm[r] = mn;
      o[0][r] *= corr; o[1][r] *= corr; o[2][r] *= corr; o[3][r] *= corr;
      // Stash P (bf16) in C-layout order -> LDS row-major [16 q][64 k].
      const int m = r + (khalf << 3);
      pw[m * 64 + cn]      = f2bf(p0);
      pw[m * 64 + 16 + cn] = f2bf(p1);
      pw[m * 64 + 32 + cn] = f2bf(p2);
      pw[m * 64 + 48 + cn] = f2bf(p3);
    }
    asm volatile("s_wait_dscnt 0" ::: "memory");

    // Re-read P as A-matrix fragments (lane<->reg transpose via LDS).
    const v16bf pf0 = load_frag_a(pw, 64, lane);        // k 0..31
    const v16bf pf1 = load_frag_a(pw + 32, 64, lane);   // k 32..63

    // ---- O += P(16x64) x V(64x16) for four d-tiles (V already in regs) ---
#pragma unroll
    for (int t = 0; t < 4; ++t) {
      o[t] = wmma_bf16(pf0, vf[t][0], o[t]);
      o[t] = wmma_bf16(pf1, vf[t][1], o[t]);
    }
  }

  // Normalize and write back token-major [B,S,H*Dh] (bf16) for out-proj GEMM.
  const int b = bh >> 4;
  const int h = bh & 15;
#pragma unroll
  for (int r = 0; r < 8; ++r) {
    const float inv = 1.0f / rl[r];
    const int srow  = q0 + r + (khalf << 3);
    bf16_t* orow = Ob + (size_t)(b * SEQ + srow) * DIM + h * DH + cn;
    orow[0]  = f2bf(o[0][r] * inv);
    orow[16] = f2bf(o[1][r] * inv);
    orow[32] = f2bf(o[2][r] * inv);
    orow[48] = f2bf(o[3][r] * inv);
  }
}

// ---------------------------------------------------------------------------
extern "C" void kernel_launch(void* const* d_in, const int* in_sizes, int n_in,
                              void* d_out, int out_size, void* d_ws, size_t ws_size,
                              hipStream_t stream) {
  (void)in_sizes; (void)n_in; (void)out_size; (void)ws_size;
  const float* x  = (const float*)d_in[0];
  const float* Wq = (const float*)d_in[1];
  const float* Wk = (const float*)d_in[2];
  const float* Wv = (const float*)d_in[3];
  const float* Wo = (const float*)d_in[4];

  const size_t SZ_X = (size_t)MTOK * DIM;   // 4,194,304 elements
  const size_t SZ_W = (size_t)DIM * DIM;    // 1,048,576 elements

  char* ws = (char*)d_ws;
  bf16_t* Xb  = (bf16_t*)ws; ws += SZ_X * 2;
  bf16_t* Wqb = (bf16_t*)ws; ws += SZ_W * 2;
  bf16_t* Wkb = (bf16_t*)ws; ws += SZ_W * 2;
  bf16_t* Wvb = (bf16_t*)ws; ws += SZ_W * 2;
  bf16_t* Wob = (bf16_t*)ws; ws += SZ_W * 2;
  bf16_t* Qb  = (bf16_t*)ws; ws += SZ_X * 2;   // [B,S,H*Dh]
  bf16_t* Kb  = (bf16_t*)ws; ws += SZ_X * 2;
  bf16_t* Vb  = (bf16_t*)ws; ws += SZ_X * 2;
  bf16_t* Qr  = (bf16_t*)ws; ws += SZ_X * 2;   // [B,H,S,Dh], RoPE'd
  bf16_t* Kr  = (bf16_t*)ws; ws += SZ_X * 2;
  bf16_t* Vtb = (bf16_t*)ws; ws += SZ_X * 2;   // [B,H,Dh,S]
  bf16_t* Ab  = (bf16_t*)ws; ws += SZ_X * 2;   // attention output, token-major

  // 1) fp32 -> bf16 conversions.
  {
    int n4 = (int)(SZ_X / 4);
    cvt_f32_bf16<<<dim3((n4 + 255) / 256), dim3(256), 0, stream>>>(x, Xb, n4);
    n4 = (int)(SZ_W / 4);
    dim3 g((n4 + 255) / 256);
    cvt_f32_bf16<<<g, dim3(256), 0, stream>>>(Wq, Wqb, n4);
    cvt_f32_bf16<<<g, dim3(256), 0, stream>>>(Wk, Wkb, n4);
    cvt_f32_bf16<<<g, dim3(256), 0, stream>>>(Wv, Wvb, n4);
    cvt_f32_bf16<<<g, dim3(256), 0, stream>>>(Wo, Wob, n4);
  }

  // 2) QKV projections: [4096,1024] = Xb @ W^T.
  {
    dim3 g(DIM / 256, MTOK / 128);          // (4, 32)
    gemm_xwT<false><<<g, dim3(256), 0, stream>>>(Xb, Wqb, (void*)Qb, MTOK, DIM, DIM);
    gemm_xwT<false><<<g, dim3(256), 0, stream>>>(Xb, Wkb, (void*)Kb, MTOK, DIM, DIM);
    gemm_xwT<false><<<g, dim3(256), 0, stream>>>(Xb, Wvb, (void*)Vb, MTOK, DIM, DIM);
  }

  // 3) RoPE + relayout Q,K; transpose V.
  {
    const int nrope = BATCH * NH * SEQ * 32;      // 2^21
    rope_relayout<<<dim3(nrope / 256), dim3(256), 0, stream>>>(Qb, Qr);
    rope_relayout<<<dim3(nrope / 256), dim3(256), 0, stream>>>(Kb, Kr);
    const int nvt = BATCH * NH * SEQ * DH;        // 2^22
    v_transpose<<<dim3(nvt / 256), dim3(256), 0, stream>>>(Vb, Vtb);
  }

  // 4) Flash attention.
  flash_attn<<<dim3(BATCH * NH, SEQ / 64), dim3(128), 0, stream>>>(Qr, Kr, Vtb, Ab);

  // 5) Output projection, f32 result straight into d_out.
  {
    dim3 g(DIM / 256, MTOK / 128);          // (4, 32)
    gemm_xwT<true><<<g, dim3(256), 0, stream>>>(Ab, Wob, d_out, MTOK, DIM, DIM);
  }
}
